// BiLSTM_CRF_MODIFY_PARALLEL_75187697483880
// MI455X (gfx1250) — matrix operand order repaired and checked by reading, not compiled, not run on hardware
//
#include <hip/hip_runtime.h>
#include <hip/hip_bf16.h>

// ---------------------------------------------------------------------------
// BiLSTM-CRF for MI455X (gfx1250, wave32, WMMA 16x16x32 f16->f32, TDM staging)
//   B=64, T=512, V=21128, E=256, H=512, K=24
// ---------------------------------------------------------------------------

typedef __attribute__((ext_vector_type(16))) _Float16 v16h;
typedef __attribute__((ext_vector_type(8)))  _Float16 v8h_t;
typedef __attribute__((ext_vector_type(8)))  float    v8f;
typedef unsigned int u32x4_t __attribute__((ext_vector_type(4)));
typedef int          i32x4_t __attribute__((ext_vector_type(4)));
typedef int          i32x8_t __attribute__((ext_vector_type(8)));

#define Bb      64
#define Tt      512
#define Ee      256
#define Hh      512
#define Kk      24
#define FOURH   2048
#define TWOH    1024
#define NEGV    (-10000.0f)
#define STARTK  22
#define STOPK   23

// LDS tile geometry: rows of 32 halves (64B) + 16B pad -> 40-half stride
#define LDSS    40
#define GEMM_BM 128
#define GEMM_BN 64

#if defined(__HIP_DEVICE_COMPILE__) && defined(__has_builtin)
#  if __has_builtin(__builtin_amdgcn_tensor_load_to_lds) && \
      __has_builtin(__builtin_amdgcn_s_wait_tensorcnt)
#    define USE_TDM 1
#  else
#    define USE_TDM 0
#  endif
#else
#  define USE_TDM 0
#endif

__device__ __forceinline__ float sigm_f(float x) {
  return 1.0f / (1.0f + __expf(-x));
}
__device__ __forceinline__ float tanh_f(float x) {
  float e = __expf(2.0f * x);
  return (e - 1.0f) / (e + 1.0f);
}

// ---------------- fragment loaders (row-major source, stride ld halves) -----
// A-frag (16x32 f16): lane l: row m0+(l&15); elems = two 16B runs at
// K = kb + (l>>4)*8 and K = kb + 16 + (l>>4)*8.
__device__ __forceinline__ v16h load_a_frag(const _Float16* __restrict__ A, int ld,
                                            int m0, int kb, int lane) {
  int r  = m0 + (lane & 15);
  int ko = kb + ((lane >> 4) << 3);
  const v8h_t* p0 = (const v8h_t*)(A + (size_t)r * ld + ko);
  const v8h_t* p1 = (const v8h_t*)(A + (size_t)r * ld + ko + 16);
  v8h_t a0 = *p0, a1 = *p1;
  v16h v;
#pragma unroll
  for (int i = 0; i < 8; ++i) { v[i] = a0[i]; v[8 + i] = a1[i]; }
  return v;
}

// B-frag (32x16 f16), B = W^T with W row-major [N][ld]:
// lane l: N-row n0+(l&15); elems = two 16B runs at
// K = kb + (l>>4)*8 and K = kb + 16 + (l>>4)*8.
__device__ __forceinline__ v16h load_b_frag(const _Float16* __restrict__ W, int ld,
                                            int n0, int kb, int lane) {
  int r  = n0 + (lane & 15);
  int ko = kb + ((lane >> 4) << 3);
  const v8h_t* p0 = (const v8h_t*)(W + (size_t)r * ld + ko);
  const v8h_t* p1 = (const v8h_t*)(W + (size_t)r * ld + ko + 16);
  v8h_t a0 = *p0, a1 = *p1;
  v16h v;
#pragma unroll
  for (int i = 0; i < 8; ++i) { v[i] = a0[i]; v[8 + i] = a1[i]; }
  return v;
}

// ---------------- TDM: 2D tile (tile_w halves x tile_h rows) -> LDS ---------
#if USE_TDM
__device__ __forceinline__ void tdm_load_2d(unsigned lds_byte_off, const void* gp,
                                            unsigned tile_w, unsigned tile_h,
                                            unsigned row_stride /*halves*/) {
  unsigned long long ga = (unsigned long long)(__UINTPTR_TYPE__)gp;
  u32x4_t g0;
  g0[0] = 1u;                                   // count=1, user descriptor
  g0[1] = lds_byte_off;                         // lds_addr
  g0[2] = (unsigned)(ga & 0xFFFFFFFFull);       // global_addr[31:0]
  g0[3] = (unsigned)((ga >> 32) & 0x1FFFFFFull) | (2u << 30);  // addr[56:32]|type=2
  i32x8_t g1;
  // data_size=1 (2B) | pad_enable | pad_interval=3 (16 DW = 64B) | pad_amount=3 (4 DW = 16B)
  g1[0] = (int)((1u << 16) | (1u << 20) | (3u << 22) | (3u << 25));
  g1[1] = (int)((tile_w & 0xFFFFu) << 16);                    // tensor_dim0[15:0]
  g1[2] = (int)(((tile_w >> 16) & 0xFFFFu) |
                ((tile_h & 0xFFFFu) << 16));                  // dim0[31:16] | dim1[15:0]
  g1[3] = (int)(((tile_h >> 16) & 0xFFFFu) |
                ((tile_w & 0xFFFFu) << 16));                  // dim1[31:16] | tile_dim0
  g1[4] = (int)(tile_h & 0xFFFFu);                            // tile_dim1 | tile_dim2=0
  g1[5] = (int)row_stride;                                    // tensor_dim0_stride[31:0]
  g1[6] = 0;                                                  // stride hi | dim1_stride lo
  g1[7] = 0;
  i32x4_t z4 = {0, 0, 0, 0};
  i32x8_t z8 = {0, 0, 0, 0, 0, 0, 0, 0};
  // clang-23 / therock-10.0 6-arg form: (g0, g1, g2, g3, g4, cpol)
  __builtin_amdgcn_tensor_load_to_lds(g0, g1, z4, z4, z8, 0);
}
#endif

// cooperative fallback staging: rows x 32 halves tile -> LDS with LDSS stride
__device__ __forceinline__ void stage_fallback(_Float16* dst, const _Float16* src,
                                               int rows, int srcLd) {
  int chunks = rows * 4;  // 4x16B per 64B row
  for (int c = threadIdx.x; c < chunks; c += blockDim.x) {
    int row = c >> 2, q = c & 3;
    *(v8h_t*)(dst + row * LDSS + q * 8) = *(const v8h_t*)(src + (size_t)row * srcLd + q * 8);
  }
}

// ---------------------------------------------------------------------------
// Tiled WMMA GEMM with TDM->LDS double buffering:
//   out16[M][ldo] = A[M][Kd] * W^T + bias   (xg precompute, f16 out)
// Block: 256 thr (8 waves), tile 128x64; wave = 2x2 of 16x16 WMMA tiles.
// ---------------------------------------------------------------------------
__global__ void wmma_gemm_tiled_kernel(const _Float16* __restrict__ A, int Kd,
                                       const _Float16* __restrict__ W,
                                       const float* __restrict__ bias,
                                       _Float16* __restrict__ out16, int ldo) {
  __shared__ _Float16 sA[2][GEMM_BM * LDSS];
  __shared__ _Float16 sB[2][GEMM_BN * LDSS];

  int wave = threadIdx.x >> 5;
  int lane = threadIdx.x & 31;
  int waveM = wave & 3;        // 0..3 -> 32-row slice
  int waveN = wave >> 2;       // 0..1 -> 32-col slice
  int blockN = blockIdx.x, blockM = blockIdx.y;

  const _Float16* aBase = A + (size_t)blockM * GEMM_BM * Kd;
  const _Float16* bBase = W + (size_t)blockN * GEMM_BN * Kd;

  auto stage = [&](int s, int kb) {
#if USE_TDM
    if ((threadIdx.x >> 5) == 0) {
      tdm_load_2d((unsigned)(__UINTPTR_TYPE__)&sA[s][0], aBase + kb, 32, GEMM_BM, (unsigned)Kd);
      tdm_load_2d((unsigned)(__UINTPTR_TYPE__)&sB[s][0], bBase + kb, 32, GEMM_BN, (unsigned)Kd);
    }
#else
    stage_fallback(&sA[s][0], aBase + kb, GEMM_BM, Kd);
    stage_fallback(&sB[s][0], bBase + kb, GEMM_BN, Kd);
#endif
  };
  auto stage_wait = [&]() {
#if USE_TDM
    if ((threadIdx.x >> 5) == 0) __builtin_amdgcn_s_wait_tensorcnt(0);
#endif
    __syncthreads();
  };

  v8f acc00 = {}, acc01 = {}, acc10 = {}, acc11 = {};
  int s = 0;
  stage(0, 0);
  stage_wait();

  int nk = Kd >> 5;
  for (int kc = 0; kc < nk; ++kc) {
    const _Float16* sAp = &sA[s][0];
    const _Float16* sBp = &sB[s][0];
    v16h a0 = load_a_frag(sAp, LDSS, waveM * 32 + 0,  0, lane);
    v16h a1 = load_a_frag(sAp, LDSS, waveM * 32 + 16, 0, lane);
    v16h b0 = load_b_frag(sBp, LDSS, waveN * 32 + 0,  0, lane);
    v16h b1 = load_b_frag(sBp, LDSS, waveN * 32 + 16, 0, lane);
    if (kc + 1 < nk) stage(s ^ 1, (kc + 1) << 5);   // overlap next DMA with math
    acc00 = __builtin_amdgcn_wmma_f32_16x16x32_f16(false, a0, false, b0, (short)0, acc00, false, false);
    acc01 = __builtin_amdgcn_wmma_f32_16x16x32_f16(false, a0, false, b1, (short)0, acc01, false, false);
    acc10 = __builtin_amdgcn_wmma_f32_16x16x32_f16(false, a1, false, b0, (short)0, acc10, false, false);
    acc11 = __builtin_amdgcn_wmma_f32_16x16x32_f16(false, a1, false, b1, (short)0, acc11, false, false);
    if (kc + 1 < nk) stage_wait();
    s ^= 1;
  }

  int hi = lane >> 4;
  int colBase = blockN * GEMM_BN + waveN * 32 + (lane & 15);
  int rowBase = blockM * GEMM_BM + waveM * 32 + 8 * hi;
#pragma unroll
  for (int ti = 0; ti < 2; ++ti) {
#pragma unroll
    for (int tj = 0; tj < 2; ++tj) {
      const v8f& acc = (ti == 0) ? (tj == 0 ? acc00 : acc01) : (tj == 0 ? acc10 : acc11);
      int colg = colBase + tj * 16;
      float bv = bias[colg];
#pragma unroll
      for (int j = 0; j < 8; ++j) {
        int rg = rowBase + ti * 16 + j;
        out16[(size_t)rg * ldo + colg] = (_Float16)(acc[j] + bv);
      }
    }
  }
}

// ---------------------------------------------------------------------------
// Simple WMMA GEMM (one 16x16 tile/wave) for the small output projection.
// ---------------------------------------------------------------------------
__global__ void wmma_gemm_bias_kernel(const _Float16* __restrict__ A, int Kd,
                                      const _Float16* __restrict__ W,
                                      const float* __restrict__ bias, int biasN,
                                      float* __restrict__ out32,
                                      int ldo, int nTilesN, int nTilesTotal) {
  int wave = threadIdx.x >> 5;
  int lane = threadIdx.x & 31;
  int tile = blockIdx.x * (blockDim.x >> 5) + wave;
  if (tile >= nTilesTotal) return;
  int mt = tile / nTilesN, nt = tile % nTilesN;
  int m0 = mt << 4, n0 = nt << 4;

  v8f acc = {};
  v16h a_c = load_a_frag(A, Kd, m0, 0, lane);
  v16h b_c = load_b_frag(W, Kd, n0, 0, lane);
  for (int kb = 0; kb < Kd; kb += 32) {
    v16h a_n = a_c, b_n = b_c;
    if (kb + 32 < Kd) {
      a_n = load_a_frag(A, Kd, m0, kb + 32, lane);
      b_n = load_b_frag(W, Kd, n0, kb + 32, lane);
    }
    acc = __builtin_amdgcn_wmma_f32_16x16x32_f16(false, a_c, false, b_c,
                                                 (short)0, acc, false, false);
    a_c = a_n; b_c = b_n;
  }

  int col = n0 + (lane & 15);
  float bv = (col < biasN) ? bias[col] : 0.0f;
  int hi = lane >> 4;
#pragma unroll
  for (int j = 0; j < 8; ++j) {
    int r = m0 + j + 8 * hi;
    out32[(size_t)r * ldo + col] = acc[j] + bv;
  }
}

// ---------------------------------------------------------------------------
// One LSTM timestep (register-pipelined): G = xg[:,t,:] + h_prev @ W_hh^T,
// fused cell. 128 waves = 4 M-tiles x 32 H-tiles; wave owns i/f/g/o tiles.
// ---------------------------------------------------------------------------
__global__ void lstm_step_kernel(const _Float16* __restrict__ hprev,  // [64][512]
                                 const _Float16* __restrict__ whh,    // [2048][512]
                                 const _Float16* __restrict__ xg,     // [B*T][2048]
                                 float* __restrict__ cstate,          // [64][512]
                                 _Float16* __restrict__ hcur,         // [64][512]
                                 _Float16* __restrict__ hout,         // [B*T][houtLd]
                                 int houtLd, int colOff, int t) {
  int wave = threadIdx.x >> 5;
  int lane = threadIdx.x & 31;
  int tile = blockIdx.x * (blockDim.x >> 5) + wave;
  if (tile >= 128) return;
  int mt = tile >> 5, nt = tile & 31;
  int m0 = mt << 4, n0 = nt << 4;
  int hi  = lane >> 4;
  int col = n0 + (lane & 15);

  v8f ci = {}, cf = {}, cg = {}, co = {};
#pragma unroll
  for (int j = 0; j < 8; ++j) {
    int r = m0 + j + 8 * hi;
    size_t xrow = ((size_t)r * Tt + t) * FOURH;
    ci[j] = (float)xg[xrow + 0 * Hh + col];
    cf[j] = (float)xg[xrow + 1 * Hh + col];
    cg[j] = (float)xg[xrow + 2 * Hh + col];
    co[j] = (float)xg[xrow + 3 * Hh + col];
  }

  // software pipeline: fetch chunk k+1 while issuing WMMAs of chunk k
  v16h a_c  = load_a_frag(hprev, Hh, m0, 0, lane);
  v16h bi_c = load_b_frag(whh, Hh, n0 + 0 * Hh, 0, lane);
  v16h bf_c = load_b_frag(whh, Hh, n0 + 1 * Hh, 0, lane);
  v16h bg_c = load_b_frag(whh, Hh, n0 + 2 * Hh, 0, lane);
  v16h bo_c = load_b_frag(whh, Hh, n0 + 3 * Hh, 0, lane);
  for (int kb = 0; kb < Hh; kb += 32) {
    v16h a_n = a_c, bi_n = bi_c, bf_n = bf_c, bg_n = bg_c, bo_n = bo_c;
    if (kb + 32 < Hh) {
      a_n  = load_a_frag(hprev, Hh, m0, kb + 32, lane);
      bi_n = load_b_frag(whh, Hh, n0 + 0 * Hh, kb + 32, lane);
      bf_n = load_b_frag(whh, Hh, n0 + 1 * Hh, kb + 32, lane);
      bg_n = load_b_frag(whh, Hh, n0 + 2 * Hh, kb + 32, lane);
      bo_n = load_b_frag(whh, Hh, n0 + 3 * Hh, kb + 32, lane);
    }
    ci = __builtin_amdgcn_wmma_f32_16x16x32_f16(false, a_c, false, bi_c, (short)0, ci, false, false);
    cf = __builtin_amdgcn_wmma_f32_16x16x32_f16(false, a_c, false, bf_c, (short)0, cf, false, false);
    cg = __builtin_amdgcn_wmma_f32_16x16x32_f16(false, a_c, false, bg_c, (short)0, cg, false, false);
    co = __builtin_amdgcn_wmma_f32_16x16x32_f16(false, a_c, false, bo_c, (short)0, co, false, false);
    a_c = a_n; bi_c = bi_n; bf_c = bf_n; bg_c = bg_n; bo_c = bo_n;
  }

#pragma unroll
  for (int j = 0; j < 8; ++j) {
    int r = m0 + j + 8 * hi;
    float iv = sigm_f(ci[j]);
    float fv = sigm_f(cf[j]);
    float gv = tanh_f(cg[j]);
    float ov = sigm_f(co[j]);
    float cold = cstate[(size_t)r * Hh + col];
    float cn = fv * cold + iv * gv;
    cstate[(size_t)r * Hh + col] = cn;
    float h = ov * tanh_f(cn);
    hcur[(size_t)r * Hh + col] = (_Float16)h;
    hout[((size_t)r * Tt + t) * houtLd + colOff + col] = (_Float16)h;
  }
}

// ---------------------------------------------------------------------------
// Elementwise helpers
// ---------------------------------------------------------------------------
__global__ void f32_to_f16_kernel(const float* __restrict__ src,
                                  _Float16* __restrict__ dst, int n) {
  int i = blockIdx.x * blockDim.x + threadIdx.x;
  if (i < n) dst[i] = (_Float16)src[i];
}

__global__ void zero_u32_kernel(unsigned int* __restrict__ p, int n) {
  int i = blockIdx.x * blockDim.x + threadIdx.x;
  if (i < n) p[i] = 0u;
}

__global__ void pad_wout_kernel(const float* __restrict__ w,
                                _Float16* __restrict__ dst) {
  int i = blockIdx.x * blockDim.x + threadIdx.x;
  if (i >= 32 * TWOH) return;
  int row = i >> 10, col = i & 1023;
  dst[i] = (row < Kk) ? (_Float16)w[row * TWOH + col] : (_Float16)0.0f;
}

__global__ void gather_embed_kernel(const int* __restrict__ sent,
                                    const float* __restrict__ embed,
                                    _Float16* __restrict__ x16, int total) {
  int i = blockIdx.x * blockDim.x + threadIdx.x;
  if (i >= total) return;
  int row = i >> 8;   // Ee = 256
  int e   = i & 255;
  int tok = sent[row];
  x16[i] = (_Float16)embed[(size_t)tok * Ee + e];
}

// ---------------------------------------------------------------------------
// Viterbi: one wave per sentence; lane = tag (K=24 <= 32).
// ---------------------------------------------------------------------------
__global__ void viterbi_kernel(const float* __restrict__ feats,  // [B*T][32]
                               const float* __restrict__ trans,  // [24][24]
                               unsigned char* __restrict__ bp,   // [B*T][32]
                               float* __restrict__ out) {        // [64 + B*T]
  int b = blockIdx.x;
  int lane = threadIdx.x;  // 0..31
  __shared__ float sv[32];
  __shared__ float str[Kk * 32];

  for (int i = lane; i < Kk * Kk; i += 32) {
    int r = i / Kk, c = i % Kk;
    str[r * 32 + c] = trans[i];
  }
  sv[lane] = (lane == STARTK) ? 0.0f : NEGV;
  __syncthreads();

  for (int t = 0; t < Tt; ++t) {
    float best = -3.0e38f;
    int bpi = 0;
    if (lane < Kk) {
      for (int p = 0; p < Kk; ++p) {
        float s = sv[p] + str[lane * 32 + p];
        if (s > best) { best = s; bpi = p; }
      }
      best += feats[((size_t)b * Tt + t) * 32 + lane];
    }
    __syncthreads();
    sv[lane] = (lane < Kk) ? best : NEGV;
    bp[((size_t)b * Tt + t) * 32 + lane] = (unsigned char)bpi;
    __syncthreads();
  }

  if (lane == 0) {
    float best = -3.0e38f;
    int tag = 0;
    for (int k = 0; k < Kk; ++k) {
      float s = sv[k] + str[STOPK * 32 + k];
      if (s > best) { best = s; tag = k; }
    }
    out[b] = best;
    for (int t = Tt - 1; t >= 0; --t) {
      out[Bb + (size_t)b * Tt + t] = (float)tag;
      tag = bp[((size_t)b * Tt + t) * 32 + tag];
    }
  }
}

// ---------------------------------------------------------------------------
// Host orchestration
// ---------------------------------------------------------------------------
extern "C" void kernel_launch(void* const* d_in, const int* in_sizes, int n_in,
                              void* d_out, int out_size, void* d_ws, size_t ws_size,
                              hipStream_t stream) {
  const int*   sentence = (const int*)  d_in[0];
  const float* embed    = (const float*)d_in[1];
  const float* w_ih_l0f = (const float*)d_in[2];
  const float* w_hh_l0f = (const float*)d_in[3];
  const float* b_l0f    = (const float*)d_in[4];
  const float* w_ih_l0b = (const float*)d_in[5];
  const float* w_hh_l0b = (const float*)d_in[6];
  const float* b_l0b    = (const float*)d_in[7];
  const float* w_ih_l1f = (const float*)d_in[8];
  const float* w_hh_l1f = (const float*)d_in[9];
  const float* b_l1f    = (const float*)d_in[10];
  const float* w_ih_l1b = (const float*)d_in[11];
  const float* w_hh_l1b = (const float*)d_in[12];
  const float* b_l1b    = (const float*)d_in[13];
  const float* w_out    = (const float*)d_in[14];
  const float* b_out    = (const float*)d_in[15];
  const float* trans    = (const float*)d_in[16];
  float* out = (float*)d_out;

  char* base = (char*)d_ws;
  size_t off = 0;
  auto alloc = [&](size_t bytes) -> char* {
    char* r = base + off;
    off = (off + bytes + 255) & ~(size_t)255;
    return r;
  };
  const size_t BT = (size_t)Bb * Tt;                 // 32768
  _Float16* x16  = (_Float16*)alloc(BT * Ee * 2);
  _Float16* h0   = (_Float16*)alloc(BT * TWOH * 2);
  _Float16* h1   = (_Float16*)alloc(BT * TWOH * 2);
  _Float16* xgF  = (_Float16*)alloc(BT * FOURH * 2);
  _Float16* xgB  = (_Float16*)alloc(BT * FOURH * 2);
  _Float16* wih_l0f16 = (_Float16*)alloc((size_t)FOURH * Ee * 2);
  _Float16* wih_l0b16 = (_Float16*)alloc((size_t)FOURH * Ee * 2);
  _Float16* whh_l0f16 = (_Float16*)alloc((size_t)FOURH * Hh * 2);
  _Float16* whh_l0b16 = (_Float16*)alloc((size_t)FOURH * Hh * 2);
  _Float16* wih_l1f16 = (_Float16*)alloc((size_t)FOURH * TWOH * 2);
  _Float16* wih_l1b16 = (_Float16*)alloc((size_t)FOURH * TWOH * 2);
  _Float16* whh_l1f16 = (_Float16*)alloc((size_t)FOURH * Hh * 2);
  _Float16* whh_l1b16 = (_Float16*)alloc((size_t)FOURH * Hh * 2);
  _Float16* wout16    = (_Float16*)alloc((size_t)32 * TWOH * 2);
  _Float16* hzero = (_Float16*)alloc((size_t)Bb * Hh * 2);
  _Float16* hp0f  = (_Float16*)alloc((size_t)Bb * Hh * 2);
  _Float16* hp1f  = (_Float16*)alloc((size_t)Bb * Hh * 2);
  _Float16* hp0b  = (_Float16*)alloc((size_t)Bb * Hh * 2);
  _Float16* hp1b  = (_Float16*)alloc((size_t)Bb * Hh * 2);
  float* c_f   = (float*)alloc((size_t)Bb * Hh * 4);
  float* c_b   = (float*)alloc((size_t)Bb * Hh * 4);
  float* feats = (float*)alloc(BT * 32 * 4);
  unsigned char* bp = (unsigned char*)alloc(BT * 32);
  (void)ws_size; (void)n_in; (void)in_sizes; (void)out_size;

  auto cvt = [&](const float* s, _Float16* d, int n) {
    f32_to_f16_kernel<<<(n + 255) / 256, 256, 0, stream>>>(s, d, n);
  };
  auto zero = [&](void* p, size_t bytes) {
    int n = (int)(bytes / 4);
    zero_u32_kernel<<<(n + 255) / 256, 256, 0, stream>>>((unsigned int*)p, n);
  };

  cvt(w_ih_l0f, wih_l0f16, FOURH * Ee);
  cvt(w_ih_l0b, wih_l0b16, FOURH * Ee);
  cvt(w_hh_l0f, whh_l0f16, FOURH * Hh);
  cvt(w_hh_l0b, whh_l0b16, FOURH * Hh);
  cvt(w_ih_l1f, wih_l1f16, FOURH * TWOH);
  cvt(w_ih_l1b, wih_l1b16, FOURH * TWOH);
  cvt(w_hh_l1f, whh_l1f16, FOURH * Hh);
  cvt(w_hh_l1b, whh_l1b16, FOURH * Hh);
  pad_wout_kernel<<<(32 * TWOH + 255) / 256, 256, 0, stream>>>(w_out, wout16);
  zero(hzero, (size_t)Bb * Hh * 2);

  {
    int total = (int)(BT * Ee);
    gather_embed_kernel<<<(total + 255) / 256, 256, 0, stream>>>(sentence, embed, x16, total);
  }

  // tiled TDM-staged GEMM for the four big input-gate precomputes
  auto gemm_xg = [&](const _Float16* A, int Kd, const _Float16* W,
                     const float* bias, _Float16* o16) {
    dim3 g(FOURH / GEMM_BN, (unsigned)(BT / GEMM_BM));  // 32 x 256
    wmma_gemm_tiled_kernel<<<g, 256, 0, stream>>>(A, Kd, W, bias, o16, FOURH);
  };

  auto run_dir = [&](const _Float16* whh16, const _Float16* xg, float* cst,
                     _Float16* hpA, _Float16* hpB, _Float16* hout, int colOff,
                     bool reverse) {
    zero(cst, (size_t)Bb * Hh * 4);
    const _Float16* hp = hzero;
    for (int i = 0; i < Tt; ++i) {
      int t = reverse ? (Tt - 1 - i) : i;
      _Float16* hc = (i & 1) ? hpB : hpA;
      lstm_step_kernel<<<16, 256, 0, stream>>>(hp, whh16, xg, cst, hc,
                                               hout, TWOH, colOff, t);
      hp = hc;
    }
  };

  // --- layer 0 ---
  gemm_xg(x16, Ee, wih_l0f16, b_l0f, xgF);
  gemm_xg(x16, Ee, wih_l0b16, b_l0b, xgB);
  run_dir(whh_l0f16, xgF, c_f, hp0f, hp1f, h0, 0,  false);
  run_dir(whh_l0b16, xgB, c_b, hp0b, hp1b, h0, Hh, true);

  // --- layer 1 ---
  gemm_xg(h0, TWOH, wih_l1f16, b_l1f, xgF);
  gemm_xg(h0, TWOH, wih_l1b16, b_l1b, xgB);
  run_dir(whh_l1f16, xgF, c_f, hp0f, hp1f, h1, 0,  false);
  run_dir(whh_l1b16, xgB, c_b, hp0b, hp1b, h1, Hh, true);

  // --- output projection -> feats [B*T][32] f32 ---
  {
    int total = (int)(BT / 16) * 2;
    wmma_gemm_bias_kernel<<<(total + 7) / 8, 256, 0, stream>>>(
        h1, TWOH, wout16, b_out, Kk, feats, 32, 2, total);
  }

  // --- CRF Viterbi decode ---
  viterbi_kernel<<<Bb, 32, 0, stream>>>(feats, trans, bp, out);
}
